// SparseRetriever_54391465837251
// MI455X (gfx1250) — matrix operand order repaired and checked by reading, not compiled, not run on hardware
//
#include <hip/hip_runtime.h>

#define B_ROWS   4096
#define NA_ROWS  8192
#define NDB      12288      // B_ROWS + NA_ROWS
#define F_DIM    512
#define K2       1024       // 2*F (re || im)
#define TOPK     32
#define TILE_N   128        // 8 waves * 16 cols
#define MBLK     32         // query rows per workgroup (2 WMMA M-tiles)
#define NSPLIT   2          // N-range splits (occupancy)
#define NRANGE   (NDB / NSPLIT)
#define EPSF     1e-7f

typedef __bf16 bf16;
typedef __attribute__((ext_vector_type(8)))  __bf16 v8bf;
typedef __attribute__((ext_vector_type(16))) __bf16 v16bf;
typedef __attribute__((ext_vector_type(8)))  float  v8f;

__device__ __forceinline__ unsigned short f2bf_rne(float f) {
    unsigned u = __float_as_uint(f);
    u += 0x7FFFu + ((u >> 16) & 1u);
    return (unsigned short)(u >> 16);
}

// ---- Kernel 1: L2-normalize db rows (re||im) -> bf16 [NDB, K2] -------------
__global__ __launch_bounds__(256)
void sr_normalize_kernel(const float* __restrict__ xr, const float* __restrict__ xi,
                         const float* __restrict__ ar, const float* __restrict__ ai,
                         unsigned short* __restrict__ dbbf)
{
    const int r   = blockIdx.x;
    const int tid = threadIdx.x;
    const float* re = (r < B_ROWS) ? (xr + (size_t)r * F_DIM)
                                   : (ar + (size_t)(r - B_ROWS) * F_DIM);
    const float* im = (r < B_ROWS) ? (xi + (size_t)r * F_DIM)
                                   : (ai + (size_t)(r - B_ROWS) * F_DIM);
    float ss = 0.f;
    for (int f = tid; f < F_DIM; f += 256) {
        float a = re[f], b = im[f];
        ss += a * a + b * b;
    }
    for (int off = 16; off > 0; off >>= 1) ss += __shfl_down(ss, off, 32);
    __shared__ float red[8];
    __shared__ float sInv;
    const int wave = tid >> 5, lane = tid & 31;
    if (lane == 0) red[wave] = ss;
    __syncthreads();
    if (tid == 0) {
        float t = 0.f;
        for (int i = 0; i < 8; ++i) t += red[i];
        sInv = rsqrtf(t);
    }
    __syncthreads();
    const float inv = sInv;
    unsigned short* row = dbbf + (size_t)r * K2;
    for (int f = tid; f < F_DIM; f += 256) {
        row[f]         = f2bf_rne(re[f] * inv);
        row[F_DIM + f] = f2bf_rne(im[f] * inv);
    }
}

// ---- Kernel 2: fused bf16-WMMA GEMM (q @ db^T) + running top-32 ------------
// Grid: (B_ROWS/MBLK, NSPLIT) x 256 threads. Block handles MBLK=32 query rows
// (two 16-row WMMA M-tiles sharing each B fragment -> half the L2 B traffic)
// over an NRANGE=6144 column strip, emitting a partial top-32 per row.
__global__ __launch_bounds__(256)
void sr_gemm_topk_kernel(const unsigned short* __restrict__ dbbf_us,
                         float* __restrict__ pv, int* __restrict__ pi)
{
    __shared__ bf16  As[MBLK * K2];               // 64 KB: A block, row-major
    __shared__ float scores[MBLK][TILE_N + 4];    // padded vs. bank conflicts
    __shared__ float topv[MBLK][TOPK];
    __shared__ int   topi[MBLK][TOPK];

    const bf16* __restrict__ db = (const bf16*)dbbf_us;
    const int tid  = threadIdx.x;
    const int wave = tid >> 5;
    const int lane = tid & 31;
    const int half = lane >> 4;     // K-half selector per WMMA operand layout
    const int l15  = lane & 15;
    const int row_base = blockIdx.x * MBLK;
    const int ns      = blockIdx.y;
    const int n_begin = ns * NRANGE;
    const int n_end   = n_begin + NRANGE;

    // Stage the 32x1024 bf16 A block into LDS (4096 uint4's, 16 per thread).
    {
        const uint4* src = (const uint4*)(dbbf_us + (size_t)row_base * K2);
        uint4* dst = (uint4*)As;
        for (int i = tid; i < (MBLK * K2) / 8; i += 256) dst[i] = src[i];
    }
    for (int i = tid; i < MBLK * TOPK; i += 256) {
        topv[i / TOPK][i % TOPK] = -1e30f;
        topi[i / TOPK][i % TOPK] = 0;
    }
    __syncthreads();

    // Per-row running min of the top-32 (threads 0..31 own one row each).
    float cmin  = -1e30f;
    int   camin = 0;

    // A fragments: lane holds row M=l15; lanes 0-15 take K 0..15,
    // lanes 16-31 take K 16..31 (contiguous, matching B).
    const bf16* Alo0 = &As[l15 * K2 + half * 16];
    const bf16* Ahi0 = Alo0 + 8;
    const bf16* Alo1 = Alo0 + 16 * K2;     // rows 16..31
    const bf16* Ahi1 = Ahi0 + 16 * K2;

    for (int cb = n_begin; cb < n_end; cb += TILE_N) {
        const int n0 = cb + wave * 16 + l15;                 // db column (row of db)
        const bf16* Bp = db + (size_t)n0 * K2 + half * 16;   // lane-contiguous K

        v8f acc0 = {0.f, 0.f, 0.f, 0.f, 0.f, 0.f, 0.f, 0.f};
        v8f acc1 = {0.f, 0.f, 0.f, 0.f, 0.f, 0.f, 0.f, 0.f};
        #pragma unroll 4
        for (int ks = 0; ks < K2; ks += 32) {
            v16bf b = *(const v16bf*)(Bp + ks);
            v8bf a0lo = *(const v8bf*)(Alo0 + ks);
            v8bf a0hi = *(const v8bf*)(Ahi0 + ks);
            v8bf a1lo = *(const v8bf*)(Alo1 + ks);
            v8bf a1hi = *(const v8bf*)(Ahi1 + ks);
            v16bf a0 = __builtin_shufflevector(a0lo, a0hi,
                          0,1,2,3,4,5,6,7,8,9,10,11,12,13,14,15);
            v16bf a1 = __builtin_shufflevector(a1lo, a1hi,
                          0,1,2,3,4,5,6,7,8,9,10,11,12,13,14,15);
            acc0 = __builtin_amdgcn_wmma_f32_16x16x32_bf16(
                       false, a0, false, b, (short)0, acc0, false, false);
            acc1 = __builtin_amdgcn_wmma_f32_16x16x32_bf16(
                       false, a1, false, b, (short)0, acc1, false, false);
        }
        // C layout: VGPR r, lanes 0-15 -> M=r, lanes 16-31 -> M=r+8; N=l15.
        const int ncol = wave * 16 + l15;
        #pragma unroll
        for (int r = 0; r < 8; ++r) {
            scores[r + 8 * half][ncol]      = acc0[r];
            scores[16 + r + 8 * half][ncol] = acc1[r];
        }
        __syncthreads();

        if (tid < MBLK) {   // thread t owns query row t: replace-min top-32
            #pragma unroll 1
            for (int c = 0; c < TILE_N; ++c) {
                float s = scores[tid][c];
                if (s > cmin) {
                    topv[tid][camin] = s;
                    topi[tid][camin] = cb + c;
                    float m = topv[tid][0]; int am = 0;
                    #pragma unroll
                    for (int j = 1; j < TOPK; ++j) {
                        float v = topv[tid][j];
                        if (v < m) { m = v; am = j; }
                    }
                    cmin = m; camin = am;
                }
            }
        }
        __syncthreads();
    }

    // Emit per-split partial top-32 (value + index) for the merge pass.
    for (int i = tid; i < MBLK * TOPK; i += 256) {
        const int r = i / TOPK, j = i % TOPK;
        const size_t o = ((size_t)ns * B_ROWS + row_base + r) * TOPK + j;
        pv[o] = topv[r][j];
        pi[o] = topi[r][j];
    }
}

// ---- Kernel 2b: merge NSPLIT partial top-32 lists -> final indices ---------
__global__ __launch_bounds__(128)
void sr_merge_topk_kernel(const float* __restrict__ pv, const int* __restrict__ pi,
                          int* __restrict__ nn_idx)
{
    const int r = blockIdx.x * 128 + threadIdx.x;
    if (r >= B_ROWS) return;

    float tv[TOPK];
    int   ti[TOPK];
    #pragma unroll
    for (int j = 0; j < TOPK; ++j) {
        tv[j] = pv[(size_t)r * TOPK + j];
        ti[j] = pi[(size_t)r * TOPK + j];
    }
    float cmin = tv[0]; int camin = 0;
    #pragma unroll
    for (int j = 1; j < TOPK; ++j)
        if (tv[j] < cmin) { cmin = tv[j]; camin = j; }

    for (int s = 1; s < NSPLIT; ++s) {
        #pragma unroll 1
        for (int j = 0; j < TOPK; ++j) {
            const size_t o = ((size_t)s * B_ROWS + r) * TOPK + j;
            const float v = pv[o];
            if (v > cmin) {
                tv[camin] = v;
                ti[camin] = pi[o];
                float m = tv[0]; int am = 0;
                #pragma unroll
                for (int q = 1; q < TOPK; ++q)
                    if (tv[q] < m) { m = tv[q]; am = q; }
                cmin = m; camin = am;
            }
        }
    }
    #pragma unroll
    for (int j = 0; j < TOPK; ++j) nn_idx[(size_t)r * TOPK + j] = ti[j];
}

// ---- Kernel 3: gather neighbors, circular mean, alpha blend ----------------
__global__ __launch_bounds__(256)
void sr_aggregate_kernel(const float* __restrict__ xr, const float* __restrict__ xi,
                         const float* __restrict__ ar, const float* __restrict__ ai,
                         const int* __restrict__ nn_idx,
                         const float* __restrict__ alpha_p,
                         float* __restrict__ out)
{
    __shared__ int sidx[TOPK];
    const int b   = blockIdx.x;
    const int tid = threadIdx.x;
    if (tid < TOPK) sidx[tid] = nn_idx[(size_t)b * TOPK + tid];
    __syncthreads();

    float a = alpha_p[0];
    a = fminf(fmaxf(a, 0.f), 1.f);

    for (int f = tid; f < F_DIM; f += 256) {
        float sum_rho = 0.f, sum_c = 0.f, sum_s = 0.f;
        #pragma unroll 4
        for (int k = 0; k < TOPK; ++k) {
            const int i = sidx[k];
            float re, im;
            if (i < B_ROWS) {
                re = xr[(size_t)i * F_DIM + f];
                im = xi[(size_t)i * F_DIM + f];
            } else {
                re = ar[(size_t)(i - B_ROWS) * F_DIM + f];
                im = ai[(size_t)(i - B_ROWS) * F_DIM + f];
            }
            const float r0 = sqrtf(re * re + im * im);
            sum_rho += r0 + EPSF;
            const float inv = (r0 > 0.f) ? (1.0f / r0) : 0.f;
            sum_c += (r0 > 0.f) ? (re * inv) : 1.0f;   // cos(atan2(0,0)) = 1
            sum_s += im * inv;                          // 0 when r0 == 0
        }
        const float mr = sum_rho * (1.f / TOPK);
        const float mc = sum_c   * (1.f / TOPK);
        const float ms = sum_s   * (1.f / TOPK);
        const float nn = sqrtf(mc * mc + ms * ms);
        float cr, ci;
        if (nn > 0.f) { cr = mc / nn; ci = ms / nn; }
        else          { cr = 1.f;     ci = 0.f;     }
        const float agg_re = mr * cr;
        const float agg_im = mr * ci;
        const size_t o = (size_t)b * F_DIM + f;
        out[o]                          = (1.f - a) * xr[o] + a * agg_re;
        out[(size_t)B_ROWS * F_DIM + o] = (1.f - a) * xi[o] + a * agg_im;
    }
}

extern "C" void kernel_launch(void* const* d_in, const int* in_sizes, int n_in,
                              void* d_out, int out_size, void* d_ws, size_t ws_size,
                              hipStream_t stream) {
    (void)in_sizes; (void)n_in; (void)out_size; (void)ws_size;
    const float* xr    = (const float*)d_in[0];
    const float* xi    = (const float*)d_in[1];
    const float* ar    = (const float*)d_in[2];
    const float* ai    = (const float*)d_in[3];
    const float* alpha = (const float*)d_in[4];
    float* out = (float*)d_out;

    // Workspace layout:
    //   [bf16 db 12288x1024 : 24 MiB]
    //   [pv  NSPLIT x 4096 x 32 f32 : 1 MiB]
    //   [pi  NSPLIT x 4096 x 32 i32 : 1 MiB]
    //   [nn_idx 4096 x 32 i32 : 0.5 MiB]
    char* ws = (char*)d_ws;
    unsigned short* dbbf = (unsigned short*)ws;
    size_t off = (size_t)NDB * K2 * sizeof(unsigned short);
    float* pv = (float*)(ws + off);           off += (size_t)NSPLIT * B_ROWS * TOPK * sizeof(float);
    int*   pi = (int*)(ws + off);             off += (size_t)NSPLIT * B_ROWS * TOPK * sizeof(int);
    int*   nn_idx = (int*)(ws + off);

    sr_normalize_kernel<<<NDB, 256, 0, stream>>>(xr, xi, ar, ai, dbbf);
    sr_gemm_topk_kernel<<<dim3(B_ROWS / MBLK, NSPLIT), 256, 0, stream>>>(dbbf, pv, pi);
    sr_merge_topk_kernel<<<(B_ROWS + 127) / 128, 128, 0, stream>>>(pv, pi, nn_idx);
    sr_aggregate_kernel<<<B_ROWS, 256, 0, stream>>>(xr, xi, ar, ai, nn_idx, alpha, out);
}
